// STDuoDecoderAttn_22746146800134
// MI455X (gfx1250) — matrix operand order repaired and checked by reading, not compiled, not run on hardware
//
#include <hip/hip_runtime.h>

// Problem dims
#define T_  20
#define B_  128
#define E_  128
#define C_  512
#define H_  256
#define V_  32000
#define XK  640      // E_ + C_
#define G4  1024     // 4*H_
#define TB  (T_*B_)  // 2560

typedef __attribute__((ext_vector_type(16))) _Float16    v16h;
typedef __attribute__((ext_vector_type(8)))  float       v8f;
typedef __attribute__((ext_vector_type(4)))  unsigned int v4u;

union HFrag { v16h v; v4u q[2]; };

// ---- WMMA fragment loaders (CDNA5 16x16x32 f16 layouts, ISA 7.12.2) ----
// A (16x32, MxK), row-major source with leading dim ld (halves).
// lane l: m = l&15; lanes 0-15 hold K 0-7 & 16-23, lanes 16-31 hold K 8-15 & 24-31.
__device__ inline v16h load_a_g(const _Float16* __restrict__ p, int ld, int lane) {
  int m = lane & 15, kh = lane >> 4;
  const _Float16* r = p + (size_t)m * ld + kh * 8;
  HFrag u;
  u.q[0] = *(const v4u*)(r);        // K = kh*8 .. +7
  u.q[1] = *(const v4u*)(r + 16);   // K = 16 + kh*8 .. +7
  return u.v;
}
// B (32x16, KxN) where B[k][n] = W[n][k], W row-major [N][ld].
// lane l: n = l&15; lanes 0-15 hold K 0-7 & 16-23, lanes 16-31 hold K 8-15 & 24-31.
__device__ inline v16h load_b_g(const _Float16* __restrict__ w, int ld, int lane) {
  int n = lane & 15, kh = lane >> 4;
  const _Float16* r = w + (size_t)n * ld + kh * 8;
  HFrag u;
  u.q[0] = *(const v4u*)(r);        // K = kh*8 .. +7
  u.q[1] = *(const v4u*)(r + 16);   // K = 16 + kh*8 .. +7
  return u.v;
}
__device__ inline v8f wmma16(v16h a, v16h b, v8f c) {
  return __builtin_amdgcn_wmma_f32_16x16x32_f16(false, a, false, b, (short)0, c,
                                                false, false);
}

__device__ inline float fast_sigm(float x) { return 1.f / (1.f + __expf(-x)); }
__device__ inline float fast_tanh(float x) { return 1.f - 2.f / (__expf(2.f * x) + 1.f); }

// LDS swizzle: 16-byte chunks (8 halves) XOR'd by row&15 so that the A-fragment
// ds_load_b128 across lanes (rows m..m+15, same chunk) hits 16 distinct bank groups.
__device__ inline int swz(int row, int j) {
  int c8 = j >> 3;
  return row * H_ + (((c8 ^ (row & 15)) << 3) | (j & 7));
}

// ---- prep kernels ----
__global__ void k_cvt(const float* __restrict__ s, _Float16* __restrict__ d, int n) {
  int i = blockIdx.x * blockDim.x + threadIdx.x;
  if (i < n) d[i] = (_Float16)s[i];
}

// X[row][0:128] = input[row], X[row][128:640] = context[row % B_]
__global__ void k_build_x(const float* __restrict__ inp, const float* __restrict__ ctx,
                          _Float16* __restrict__ X) {
  int i = blockIdx.x * blockDim.x + threadIdx.x;
  if (i >= TB * XK) return;
  int row = i / XK, col = i - row * XK;
  int b = row & (B_ - 1);
  float v = (col < E_) ? inp[(size_t)row * E_ + col] : ctx[(size_t)b * C_ + (col - E_)];
  X[i] = (_Float16)v;
}

// ---- xg = X @ Wih^T + (bih + bhh), fp32 [TB][G4] ----
// grid (TB/128, G4/64), block 256 (8 waves; wave = one 16-row M tile, 4 N tiles).
// k-loop fully unrolled; B fragments batched per k so loads clause together and
// WMMA waits stagger. No launch_bounds occupancy floor: these GEMMs hide L2
// latency with intra-wave ILP, so give the allocator register headroom.
__global__ __launch_bounds__(256) void k_xg(const _Float16* __restrict__ X,
                                            const _Float16* __restrict__ Wih,
                                            const float* __restrict__ bih,
                                            const float* __restrict__ bhh,
                                            float* __restrict__ xg) {
  int lane = threadIdx.x & 31, w = threadIdx.x >> 5;
  int m0 = blockIdx.x * 128 + w * 16;
  int n0 = blockIdx.y * 64;
  v8f acc[4] = {};
  const _Float16* Xp = X + (size_t)m0 * XK;
#pragma unroll
  for (int k = 0; k < XK / 32; ++k) {
    v16h a = load_a_g(Xp + k * 32, XK, lane);
    v16h b[4];
#pragma unroll
    for (int nt = 0; nt < 4; ++nt)
      b[nt] = load_b_g(Wih + (size_t)(n0 + nt * 16) * XK + k * 32, XK, lane);
#pragma unroll
    for (int nt = 0; nt < 4; ++nt) acc[nt] = wmma16(a, b[nt], acc[nt]);
  }
  int mb = m0 + 8 * (lane >> 4);
#pragma unroll
  for (int nt = 0; nt < 4; ++nt) {
    int col = n0 + nt * 16 + (lane & 15);
    float bias = bih[col] + bhh[col];
#pragma unroll
    for (int r = 0; r < 8; ++r)
      xg[(size_t)(mb + r) * G4 + col] = acc[nt][r] + bias;
  }
}

// ---- recurrence: 2 blocks (one per LSTM), 1024 threads = 32 waves ----
// Wave w = (col-group cg = w&15 : 16 hidden columns, mh = w>>4 : batch half).
// Gates: i = N-tile cg, g = 32+cg, o = 48+cg (f-gate skipped: c0 == 0).
__global__ __launch_bounds__(1024) void k_lstm(
    const _Float16* __restrict__ Whh_p, const _Float16* __restrict__ Whh_n,
    const float* __restrict__ xg_p, const float* __restrict__ xg_n,
    const float* __restrict__ h0_p, const float* __restrict__ h0_n,
    _Float16* __restrict__ hs_p, _Float16* __restrict__ hs_n) {
  const _Float16* Whh = blockIdx.x ? Whh_n : Whh_p;
  const float*    xg  = blockIdx.x ? xg_n  : xg_p;
  const float*    h0  = blockIdx.x ? h0_n  : h0_p;
  _Float16*       hs  = blockIdx.x ? hs_n  : hs_p;

  __shared__ _Float16 hsm[B_ * H_];  // 64 KB, swizzled
  int tid = threadIdx.x, lane = tid & 31, w = tid >> 5;

  for (int i = tid; i < B_ * H_; i += 1024) {
    int row = i >> 8, j = i & (H_ - 1);
    hsm[swz(row, j)] = (_Float16)h0[j];  // h0 broadcast over batch
  }
  __syncthreads();

  int cg = w & 15, mh = w >> 4;
  int nt_i = cg, nt_g = 32 + cg, nt_o = 48 + cg;

  for (int t = 0; t < T_; ++t) {
    v8f acc0[4] = {}, acc1[4] = {}, acc2[4] = {};
#pragma unroll
    for (int k = 0; k < 8; ++k) {
      // Batch the 3 gate B fragments (global, L2-resident) ...
      v16h b0 = load_b_g(Whh + (size_t)nt_i * 16 * H_ + k * 32, H_, lane);
      v16h b1 = load_b_g(Whh + (size_t)nt_g * 16 * H_ + k * 32, H_, lane);
      v16h b2 = load_b_g(Whh + (size_t)nt_o * 16 * H_ + k * 32, H_, lane);
      // ... and the 4 A fragments (LDS, swizzled, conflict-free ds_load_b128).
      v16h a[4];
#pragma unroll
      for (int mt = 0; mt < 4; ++mt) {
        int row = mh * 64 + mt * 16 + (lane & 15);
        int kh = lane >> 4;
        int c0 = k * 4 + kh;
        HFrag u;
        u.q[0] = *(const v4u*)(hsm + row * H_ + ((c0 ^ (row & 15)) << 3));
        u.q[1] = *(const v4u*)(hsm + row * H_ + (((c0 + 2) ^ (row & 15)) << 3));
        a[mt] = u.v;
      }
#pragma unroll
      for (int mt = 0; mt < 4; ++mt) {
        acc0[mt] = wmma16(a[mt], b0, acc0[mt]);
        acc1[mt] = wmma16(a[mt], b1, acc1[mt]);
        acc2[mt] = wmma16(a[mt], b2, acc2[mt]);
      }
    }
    __syncthreads();  // all LDS reads of h(t) complete

    int jcol = cg * 16 + (lane & 15);
#pragma unroll
    for (int mt = 0; mt < 4; ++mt) {
      int mbase = mh * 64 + mt * 16 + 8 * (lane >> 4);
#pragma unroll
      for (int r = 0; r < 8; ++r) {
        int br = mbase + r;
        const float* xr = xg + (size_t)(t * B_ + br) * G4;
        float gi = acc0[mt][r] + xr[jcol];         // i-gate
        float gg = acc1[mt][r] + xr[512 + jcol];   // g-gate
        float go = acc2[mt][r] + xr[768 + jcol];   // o-gate
        float cc = fast_sigm(gi) * fast_tanh(gg);  // f*c0 == 0 (reference bug)
        float hv = fast_sigm(go) * fast_tanh(cc);
        _Float16 hh = (_Float16)hv;
        hsm[swz(br, jcol)] = hh;
        hs[(size_t)(t * B_ + br) * H_ + jcol] = hh;
      }
    }
    __syncthreads();  // h(t+1) visible to all waves
  }
}

// ---- logits = hs @ Wp^T + bp, fp32 out [TB][V] ----
// grid (TB/128, V/128), block 256: wave = one M tile x 8 N tiles, K = 256.
// All 8 B fragments of a k-step preloaded so the 16 b128 loads issue as one
// clause and the WMMA waits stagger instead of hitting 0.
__global__ __launch_bounds__(256) void k_logits(const _Float16* __restrict__ hs,
                                                const _Float16* __restrict__ Wp,
                                                const float* __restrict__ bp,
                                                float* __restrict__ out) {
  int lane = threadIdx.x & 31, w = threadIdx.x >> 5;
  int m0 = blockIdx.x * 128 + w * 16;
  int n0 = blockIdx.y * 128;
  v8f acc[8] = {};
  const _Float16* hp = hs + (size_t)m0 * H_;
#pragma unroll
  for (int k = 0; k < 8; ++k) {
    v16h a = load_a_g(hp + k * 32, H_, lane);
    v16h b[8];
#pragma unroll
    for (int nt = 0; nt < 8; ++nt)
      b[nt] = load_b_g(Wp + (size_t)(n0 + nt * 16) * H_ + k * 32, H_, lane);
#pragma unroll
    for (int nt = 0; nt < 8; ++nt) acc[nt] = wmma16(a, b[nt], acc[nt]);
  }
  int mb = m0 + 8 * (lane >> 4);
#pragma unroll
  for (int nt = 0; nt < 8; ++nt) {
    int col = n0 + nt * 16 + (lane & 15);
    float bias = bp[col];
#pragma unroll
    for (int r = 0; r < 8; ++r)
      out[(size_t)(mb + r) * V_ + col] = acc[nt][r] + bias;
  }
}

extern "C" void kernel_launch(void* const* d_in, const int* in_sizes, int n_in,
                              void* d_out, int out_size, void* d_ws, size_t ws_size,
                              hipStream_t stream) {
  (void)in_sizes; (void)n_in; (void)out_size; (void)ws_size;
  const float* inputPrev = (const float*)d_in[0];
  const float* inputNext = (const float*)d_in[1];
  const float* context   = (const float*)d_in[2];
  const float* hidden1   = (const float*)d_in[3];
  const float* hidden2   = (const float*)d_in[4];
  const float* Wih_p = (const float*)d_in[5];
  const float* Whh_p = (const float*)d_in[6];
  const float* bih_p = (const float*)d_in[7];
  const float* bhh_p = (const float*)d_in[8];
  const float* Wih_n = (const float*)d_in[9];
  const float* Whh_n = (const float*)d_in[10];
  const float* bih_n = (const float*)d_in[11];
  const float* bhh_n = (const float*)d_in[12];
  const float* Wp    = (const float*)d_in[13];
  const float* bp    = (const float*)d_in[14];
  float* out = (float*)d_out;

  // Workspace layout (all sizes multiples of 256 B; total ~50.2 MB)
  char* ws = (char*)d_ws;
  size_t o = 0;
  _Float16* Wih_p16 = (_Float16*)(ws + o); o += (size_t)G4 * XK * 2;
  _Float16* Wih_n16 = (_Float16*)(ws + o); o += (size_t)G4 * XK * 2;
  _Float16* Whh_p16 = (_Float16*)(ws + o); o += (size_t)G4 * H_ * 2;
  _Float16* Whh_n16 = (_Float16*)(ws + o); o += (size_t)G4 * H_ * 2;
  _Float16* Wp16    = (_Float16*)(ws + o); o += (size_t)V_ * H_ * 2;
  _Float16* Xp16    = (_Float16*)(ws + o); o += (size_t)TB * XK * 2;
  _Float16* Xn16    = (_Float16*)(ws + o); o += (size_t)TB * XK * 2;
  float*    xg_p    = (float*)(ws + o);    o += (size_t)TB * G4 * 4;
  float*    xg_n    = (float*)(ws + o);    o += (size_t)TB * G4 * 4;
  _Float16* hs_p16  = (_Float16*)(ws + o); o += (size_t)TB * H_ * 2;
  _Float16* hs_n16  = (_Float16*)(ws + o); o += (size_t)TB * H_ * 2;

  const int thr = 256;
  auto blks = [](size_t n, int t) { return (unsigned)((n + t - 1) / t); };

  // Stage f16 operands
  k_cvt<<<blks((size_t)G4 * XK, thr), thr, 0, stream>>>(Wih_p, Wih_p16, G4 * XK);
  k_cvt<<<blks((size_t)G4 * XK, thr), thr, 0, stream>>>(Wih_n, Wih_n16, G4 * XK);
  k_cvt<<<blks((size_t)G4 * H_, thr), thr, 0, stream>>>(Whh_p, Whh_p16, G4 * H_);
  k_cvt<<<blks((size_t)G4 * H_, thr), thr, 0, stream>>>(Whh_n, Whh_n16, G4 * H_);
  k_cvt<<<blks((size_t)V_ * H_, thr), thr, 0, stream>>>(Wp, Wp16, V_ * H_);
  k_build_x<<<blks((size_t)TB * XK, thr), thr, 0, stream>>>(inputPrev, context, Xp16);
  k_build_x<<<blks((size_t)TB * XK, thr), thr, 0, stream>>>(inputNext, context, Xn16);

  // Input-gate GEMMs (biases folded in)
  k_xg<<<dim3(TB / 128, G4 / 64), 256, 0, stream>>>(Xp16, Wih_p16, bih_p, bhh_p, xg_p);
  k_xg<<<dim3(TB / 128, G4 / 64), 256, 0, stream>>>(Xn16, Wih_n16, bih_n, bhh_n, xg_n);

  // Both LSTM recurrences (one block each, run concurrently)
  k_lstm<<<2, 1024, 0, stream>>>(Whh_p16, Whh_n16, xg_p, xg_n,
                                 hidden1, hidden2, hs_p16, hs_n16);

  // Vocab projections
  k_logits<<<dim3(TB / 128, V_ / 128), 256, 0, stream>>>(hs_p16, Wp16, bp, out);
  k_logits<<<dim3(TB / 128, V_ / 128), 256, 0, stream>>>(hs_n16, Wp16, bp,
                                                         out + (size_t)TB * V_);
}
